// ProductQuantization_12618613916102
// MI455X (gfx1250) — compile-verified, matrix operand order checked by reading, not run on hardware
//
#include <hip/hip_runtime.h>
#include <hip/hip_bf16.h>
#include <stdint.h>

// ---------------------------------------------------------------------------
// Types for CDNA5 WMMA (wave32): v_wmma_f32_16x16x32_bf16
// ---------------------------------------------------------------------------
typedef __attribute__((ext_vector_type(16))) __bf16 v16bf;
typedef __attribute__((ext_vector_type(8)))  float  v8f;

union FragBF {
    v16bf v;
    uint4 q[2];   // 32 bytes
};

__device__ __forceinline__ unsigned short f32_bf16(float f) {
    unsigned u = __builtin_bit_cast(unsigned, f);
    u += 0x7FFFu + ((u >> 16) & 1u);          // round-to-nearest-even
    return (unsigned short)(u >> 16);
}
__device__ __forceinline__ unsigned pack_bf16x2(float a, float b) {
    return (unsigned)f32_bf16(a) | ((unsigned)f32_bf16(b) << 16);
}

// ---------------------------------------------------------------------------
// f32 -> bf16 elementwise convert
// ---------------------------------------------------------------------------
__global__ void cvt_f32_bf16_kernel(const float* __restrict__ in,
                                    unsigned short* __restrict__ out, int n) {
    int i = (blockIdx.x * blockDim.x + threadIdx.x) * 4;
    if (i + 3 < n) {
        float4 f = *(const float4*)(in + i);
        uint2 o;
        o.x = pack_bf16x2(f.x, f.y);
        o.y = pack_bf16x2(f.z, f.w);
        *(uint2*)(out + i) = o;
    }
}

// ---------------------------------------------------------------------------
// Row-dot GEMM: C[i,j] = sum_{k<512} A[i*512+k] * B[j*512+k]
// A: Na rows of 512 bf16, B: Nb rows of 512 bf16. Out: f32 or bf16, ld = Nb.
// block = 256 threads (8 waves); grid = (Na/16, Nb/256); wave owns 2 col tiles.
// ---------------------------------------------------------------------------
template <bool OUT_BF16>
__global__ void __launch_bounds__(256)
gemm_rr_kernel(const unsigned short* __restrict__ A,
               const unsigned short* __restrict__ B,
               void* __restrict__ Cout, int ldc) {
    __shared__ unsigned short alds[16 * 520];
    const int tid  = threadIdx.x;
    const int wave = tid >> 5, lane = tid & 31;
    const int half = lane >> 4, l16 = lane & 15;
    const int row0 = blockIdx.x * 16;
    const int col0 = blockIdx.y * 256 + wave * 32;

    // stage A tile (16x512 bf16) into LDS, padded stride 520
    {
        int e = tid * 32;
        int r = e >> 9, c = e & 511;
        const uint4* src = (const uint4*)(A + (size_t)(row0 + r) * 512 + c);
        uint4* dst = (uint4*)(&alds[r * 520 + c]);
#pragma unroll
        for (int q = 0; q < 4; ++q) dst[q] = src[q];
    }
    __syncthreads();

    v8f acc0 = {}, acc1 = {};
    for (int kc = 0; kc < 512; kc += 32) {
        FragBF a;
        const uint4* ap = (const uint4*)(&alds[l16 * 520 + kc + half * 8]);
        a.q[0] = ap[0]; a.q[1] = ap[2];   // K = kc+{0..7,16..23} / {8..15,24..31}
        FragBF b0, b1;
        const uint4* bp0 = (const uint4*)(B + (size_t)(col0 + l16) * 512 + kc + half * 8);
        const uint4* bp1 = (const uint4*)(B + (size_t)(col0 + 16 + l16) * 512 + kc + half * 8);
        b0.q[0] = bp0[0]; b0.q[1] = bp0[2];
        b1.q[0] = bp1[0]; b1.q[1] = bp1[2];
        acc0 = __builtin_amdgcn_wmma_f32_16x16x32_bf16(false, a.v, false, b0.v,
                                                       (short)0, acc0, false, false);
        acc1 = __builtin_amdgcn_wmma_f32_16x16x32_bf16(false, a.v, false, b1.v,
                                                       (short)0, acc1, false, false);
    }

#pragma unroll
    for (int i = 0; i < 8; ++i) {
        size_t r = (size_t)(row0 + half * 8 + i);
        int c0 = col0 + l16, c1 = col0 + 16 + l16;
        if (OUT_BF16) {
            unsigned short* o = (unsigned short*)Cout;
            o[r * ldc + c0] = f32_bf16(acc0[i]);
            o[r * ldc + c1] = f32_bf16(acc1[i]);
        } else {
            float* o = (float*)Cout;
            o[r * ldc + c0] = acc0[i];
            o[r * ldc + c1] = acc1[i];
        }
    }
}

// ---------------------------------------------------------------------------
// Fused main kernel: per block, 16 rows of X.
//   1) L[16 x 2048] = Xtile @ Pt^T via WMMA (wave w owns cols [256w, 256w+256))
//   2) 8 sequential argmax steps with G-corrections, codes in LDS
//   3) results = sum of selected f32 codewords; write int32 codes
// ---------------------------------------------------------------------------
__global__ void __launch_bounds__(256)
pq_main_kernel(const float* __restrict__ X,             // [N,512] f32
               const unsigned short* __restrict__ Pt,   // [2048,512] bf16
               const float* __restrict__ G,             // [2048,2048] f32
               const float* __restrict__ CB,            // [2048,512] f32
               float* __restrict__ results,             // [N,512] f32
               int* __restrict__ codes_out) {           // [N,8] i32
    __shared__ unsigned short xlds[16 * 520];
    __shared__ int codes_lds[16][8];

    const int tid  = threadIdx.x;
    const int wave = tid >> 5, lane = tid & 31;
    const int half = lane >> 4, l16 = lane & 15;
    const size_t row0 = (size_t)blockIdx.x * 16;

    // stage X tile, converting f32 -> bf16 into LDS (stride 520)
    {
        int e = tid * 32;
        int r = e >> 9, c = e & 511;
        const float4* src = (const float4*)(X + (row0 + r) * 512 + c);
        uint4* dst = (uint4*)(&xlds[r * 520 + c]);
#pragma unroll
        for (int q = 0; q < 4; ++q) {
            float4 f0 = src[2 * q], f1 = src[2 * q + 1];
            uint4 o;
            o.x = pack_bf16x2(f0.x, f0.y);
            o.y = pack_bf16x2(f0.z, f0.w);
            o.z = pack_bf16x2(f1.x, f1.y);
            o.w = pack_bf16x2(f1.z, f1.w);
            dst[q] = o;
        }
    }
    __syncthreads();

    // ---- main GEMM: 16 accumulator tiles per wave (16 rows x 256 cols) ----
    v8f acc[16];
#pragma unroll
    for (int t = 0; t < 16; ++t) acc[t] = (v8f){};

    const int colw = wave * 256;
    for (int kc = 0; kc < 512; kc += 32) {
        FragBF a;
        const uint4* ap = (const uint4*)(&xlds[l16 * 520 + kc + half * 8]);
        a.q[0] = ap[0]; a.q[1] = ap[2];
#pragma unroll
        for (int t = 0; t < 16; ++t) {
            FragBF b;
            const uint4* bp =
                (const uint4*)(Pt + (size_t)(colw + t * 16 + l16) * 512 + kc + half * 8);
            b.q[0] = bp[0]; b.q[1] = bp[2];
            acc[t] = __builtin_amdgcn_wmma_f32_16x16x32_bf16(false, a.v, false, b.v,
                                                             (short)0, acc[t], false, false);
        }
    }
    __syncthreads();

    // ---- sequential quantization steps: wave m handles step m ----
    for (int m = 0; m < 8; ++m) {
        if (wave == m) {
#pragma unroll
            for (int i = 0; i < 8; ++i) {
                int row = half * 8 + i;
                float best = -3.4e38f;
                int   bidx = 0x7fffffff;
#pragma unroll
                for (int t = 0; t < 16; ++t) {
                    float v = acc[t][i];
                    int col = t * 16 + l16;           // 0..255 within block m
                    for (int j = 0; j < m; ++j) {
                        int cj = codes_lds[row][j];
                        v -= G[(size_t)(j * 256 + cj) * 2048 + (size_t)m * 256 + col];
                    }
                    if (v > best || (v == best && col < bidx)) { best = v; bidx = col; }
                }
                // butterfly argmax across the 16-lane half (min index on ties)
#pragma unroll
                for (int off = 8; off >= 1; off >>= 1) {
                    float ov = __shfl_xor(best, off, 32);
                    int   oi = __shfl_xor(bidx, off, 32);
                    if (ov > best || (ov == best && oi < bidx)) { best = ov; bidx = oi; }
                }
                if (l16 == 0) codes_lds[row][m] = bidx;
            }
        }
        __syncthreads();
    }

    // ---- epilogue: results = sum_m CB[m*256 + code_m], plus codes out ----
    {
        int r = tid >> 4;                  // 16 threads per row
        int cbase = (tid & 15) * 32;       // 32 cols each
        const float* p[8];
#pragma unroll
        for (int m = 0; m < 8; ++m)
            p[m] = CB + (size_t)(m * 256 + codes_lds[r][m]) * 512 + cbase;
#pragma unroll
        for (int q = 0; q < 8; ++q) {
            float4 s = *(const float4*)(p[0] + q * 4);
#pragma unroll
            for (int m = 1; m < 8; ++m) {
                float4 f = *(const float4*)(p[m] + q * 4);
                s.x += f.x; s.y += f.y; s.z += f.z; s.w += f.w;
            }
            *(float4*)(results + (row0 + r) * 512 + cbase + q * 4) = s;
        }
    }
    if (tid < 128) {
        int r = tid >> 3, m = tid & 7;
        codes_out[(row0 + r) * 8 + m] = codes_lds[r][m];
    }
}

// ---------------------------------------------------------------------------
// Launch
// ---------------------------------------------------------------------------
extern "C" void kernel_launch(void* const* d_in, const int* in_sizes, int n_in,
                              void* d_out, int out_size, void* d_ws, size_t ws_size,
                              hipStream_t stream) {
    const float* x  = (const float*)d_in[0];   // [N, 512]
    const float* cb = (const float*)d_in[1];   // [8, 256, 512] == [2048, 512]
    const float* w  = (const float*)d_in[2];   // [512, 512]

    const int D = 512, MK = 2048;              // M=8, K=256
    const int Nrows = in_sizes[0] / D;         // 131072

    char* ws = (char*)d_ws;
    unsigned short* cb_bf = (unsigned short*)(ws);              // 2 MB
    unsigned short* w_bf  = (unsigned short*)(ws + (2u << 20)); // 0.5 MB
    unsigned short* pt_bf = (unsigned short*)(ws + 2621440u);   // 2 MB
    float*          g     = (float*)(ws + 4718592u);            // 16 MB

    float* results = (float*)d_out;
    int*   codes   = (int*)((float*)d_out + (size_t)Nrows * D);

    // 1) convert CB, W to bf16
    cvt_f32_bf16_kernel<<<(MK * D / 4 + 255) / 256, 256, 0, stream>>>(cb, cb_bf, MK * D);
    cvt_f32_bf16_kernel<<<(D * D / 4 + 255) / 256, 256, 0, stream>>>(w, w_bf, D * D);
    // 2) Pt[c,d] = sum_e CB[c,e] * W[d,e]   -> bf16 [2048 x 512]
    gemm_rr_kernel<true><<<dim3(MK / 16, D / 256), 256, 0, stream>>>(cb_bf, w_bf, pt_bf, D);
    // 3) G[r,c]  = sum_d CB[r,d] * Pt[c,d]  -> f32  [2048 x 2048]
    gemm_rr_kernel<false><<<dim3(MK / 16, MK / 256), 256, 0, stream>>>(cb_bf, pt_bf, g, MK);
    // 4) fused big GEMM + argmax scan + codeword sum
    pq_main_kernel<<<Nrows / 16, 256, 0, stream>>>(x, pt_bf, g, cb, results, codes);
}
